// GCN_76484777607563
// MI455X (gfx1250) — matrix operand order repaired and checked by reading, not compiled, not run on hardware
//
#include <hip/hip_runtime.h>
#include <hip/hip_bf16.h>
#include <math.h>

typedef __attribute__((ext_vector_type(16))) _Float16 v16h;
typedef __attribute__((ext_vector_type(8)))  _Float16 v8h;
typedef __attribute__((ext_vector_type(8)))  float    v8f;

#define NFEAT  9
#define H1     32
#define H2     64
#define NGRAPH 64
#define KSTEPS 17          // K = 17*32 = 544 (16 h-weighted blocks + 1 bias/x block)

// ---------------------------------------------------------------- utilities
__global__ void k_zero(float* __restrict__ p, int n) {
    int i = blockIdx.x * blockDim.x + threadIdx.x;
    if (i < n) p[i] = 0.f;
}

// per-edge MLPs: h = relu(edge_attr * w1 + b1) for both layers; also degree count
__global__ void k_edge_mlp(const float* __restrict__ ea, const int* __restrict__ dst,
                           const float* __restrict__ w1a, const float* __restrict__ b1a,
                           const float* __restrict__ w1b, const float* __restrict__ b1b,
                           _Float16* __restrict__ he1, _Float16* __restrict__ he2,
                           float* __restrict__ deg, int E) {
    int e = blockIdx.x * blockDim.x + threadIdx.x;
    if (e >= E) return;
    float a = ea[e];
#pragma unroll
    for (int j = 0; j < 16; ++j) {
        float v1 = a * w1a[j] + b1a[j];
        float v2 = a * w1b[j] + b1b[j];
        he1[e * 16 + j] = (_Float16)(v1 > 0.f ? v1 : 0.f);
        he2[e * 16 + j] = (_Float16)(v2 > 0.f ? v2 : 0.f);
    }
    atomicAdd(&deg[dst[e]], 1.f);
}

// pack node features x [N,9] -> f16 [N,32] (zero padded)
__global__ void k_pack_x(const float* __restrict__ x, _Float16* __restrict__ xf, int N) {
    int i = blockIdx.x * blockDim.x + threadIdx.x;
    if (i >= N * 32) return;
    int n = i >> 5, c = i & 31;
    xf[i] = (c < NFEAT) ? (_Float16)x[n * NFEAT + c] : (_Float16)0.f;
}

// Build Wc in pre-swizzled WMMA B-fragment order.
// Fragment (nt,kk): 32 lanes x 16 halves; lane holds col nt*16+(lane&15);
// half t = Wc[kk*32 + (lane>=16?8:0) + (t<8?t:t+8), col]   (B table-order striping)
// Wc[k*32+i, col] = (k<16) ? w2[k, i*cout+col] : b2[i*cout+col]   (i<cin, else 0)
__device__ inline void pack_one(_Float16* __restrict__ wp, const float* __restrict__ w2,
                                const float* __restrict__ b2, int cin, int cout, int idx) {
    int frag = idx >> 9;             // /512
    int rem  = idx & 511;
    int lane = rem >> 4;
    int t    = rem & 15;
    int kk   = frag % KSTEPS;
    int nt   = frag / KSTEPS;
    int col  = nt * 16 + (lane & 15);
    int i    = ((lane >= 16) ? 8 : 0) + ((t < 8) ? t : t + 8);
    float v  = 0.f;
    if (i < cin) {
        if (kk < 16) v = w2[kk * (cin * cout) + i * cout + col];
        else         v = b2[i * cout + col];
    }
    wp[idx] = (_Float16)v;
}

__global__ void k_pack_w(const float* __restrict__ w2a, const float* __restrict__ b2a,
                         const float* __restrict__ w2b, const float* __restrict__ b2b,
                         _Float16* __restrict__ wp1, _Float16* __restrict__ wp2) {
    int idx = blockIdx.x * blockDim.x + threadIdx.x;
    const int n1 = (H1 / 16) * KSTEPS * 512;
    const int n2 = (H2 / 16) * KSTEPS * 512;
    if (idx < n1)            pack_one(wp1, w2a, b2a, NFEAT, H1, idx);
    else if (idx < n1 + n2)  pack_one(wp2, w2b, b2b, H1,    H2, idx - n1);
}

// ---------------------------------------------------------------- edge GEMM
// One wave handles 16 edges: D[16, NT*16] = Z[16,544] @ Wc[544, NT*16] via WMMA,
// then atomic-scatters rows into agg[dst].
template <int NT>
__global__ void __launch_bounds__(256)
k_edge_gemm(const _Float16* __restrict__ feat,   // [N,32] f16 padded
            const _Float16* __restrict__ hedge,  // [E,16] f16
            const _Float16* __restrict__ wcp,    // packed B fragments
            const int* __restrict__ src, const int* __restrict__ dst,
            float* __restrict__ agg, int E) {
    const int lane = threadIdx.x & 31;
    const int wave = blockIdx.x * (blockDim.x >> 5) + (threadIdx.x >> 5);
    const int e0   = wave * 16;
    if (e0 >= E) return;                         // whole-wave uniform exit
    const int r  = lane & 15;
    const int hi = lane >> 4;
    int e = e0 + r; if (e >= E) e = E - 1;

    // A-layout lane feature subsets: lanes0-15 -> i{0..7,16..23}; lanes16-31 -> i{8..15,24..31}
    const _Float16* fp = feat + (size_t)src[e] * 32;
    v8h fa = *(const v8h*)(fp + hi * 8);
    v8h fb = *(const v8h*)(fp + 16 + hi * 8);

    const _Float16* hp = hedge + (size_t)e * 16;
    v8h h0 = *(const v8h*)hp;
    v8h h1 = *(const v8h*)(hp + 8);
    _Float16 hv[16];
#pragma unroll
    for (int t = 0; t < 8; ++t) { hv[t] = h0[t]; hv[t + 8] = h1[t]; }

    v8f acc[NT];
#pragma unroll
    for (int nt = 0; nt < NT; ++nt) acc[nt] = (v8f){};

#pragma unroll
    for (int kk = 0; kk < KSTEPS; ++kk) {
        v16h a;
        if (kk < 16) {
            _Float16 s = hv[kk];
#pragma unroll
            for (int t = 0; t < 8; ++t) { a[t] = fa[t] * s; a[t + 8] = fb[t] * s; }
        } else {    // bias / passthrough block: z = feat
#pragma unroll
            for (int t = 0; t < 8; ++t) { a[t] = fa[t]; a[t + 8] = fb[t]; }
        }
#pragma unroll
        for (int nt = 0; nt < NT; ++nt) {
            v16h b = *(const v16h*)(wcp + ((size_t)(nt * KSTEPS + kk) * 32 + lane) * 16);
            acc[nt] = __builtin_amdgcn_wmma_f32_16x16x32_f16(
                false, a, false, b, (short)0, acc[nt], false, false);
        }
    }

    // C/D layout: VGPR v -> row v + hi*8, col = lane&15 (within N-tile)
    int drow[8];
#pragma unroll
    for (int v = 0; v < 8; ++v) {
        int er = e0 + hi * 8 + v;
        drow[v] = (er < E) ? dst[er] : -1;
    }
#pragma unroll
    for (int nt = 0; nt < NT; ++nt)
#pragma unroll
        for (int v = 0; v < 8; ++v)
            if (drow[v] >= 0)
                atomicAdd(&agg[(size_t)drow[v] * (NT * 16) + nt * 16 + r], acc[nt][v]);
}

// ---------------------------------------------------------------- node updates
__global__ void k_node1(const float* __restrict__ x, const float* __restrict__ agg,
                        const float* __restrict__ deg, const float* __restrict__ root,
                        const float* __restrict__ bias,
                        float* __restrict__ h, _Float16* __restrict__ hf, int N) {
    int idx = blockIdx.x * blockDim.x + threadIdx.x;
    if (idx >= N * H1) return;
    int n = idx >> 5, o = idx & 31;
    float s = bias[o] + agg[idx] / fmaxf(deg[n], 1.f);
#pragma unroll
    for (int i = 0; i < NFEAT; ++i) s += x[n * NFEAT + i] * root[i * H1 + o];
    s = (s > 0.f) ? s : (expf(s) - 1.f);     // ELU
    h[idx] = s;
    hf[idx] = (_Float16)s;
}

__global__ void k_node2(const float* __restrict__ h, const float* __restrict__ agg,
                        const float* __restrict__ deg, const int* __restrict__ batch,
                        const float* __restrict__ root, const float* __restrict__ bias,
                        float* __restrict__ node_feat, float* __restrict__ gsum,
                        float* __restrict__ gcnt, int N) {
    int idx = blockIdx.x * blockDim.x + threadIdx.x;
    if (idx >= N * H2) return;
    int n = idx >> 6, o = idx & 63;
    float s = bias[o] + agg[idx] / fmaxf(deg[n], 1.f);
#pragma unroll
    for (int i = 0; i < H1; ++i) s += h[n * H1 + i] * root[i * H2 + o];
    s = (s > 0.f) ? s : (expf(s) - 1.f);     // ELU
    node_feat[idx] = s;
    int b = batch[n];
    atomicAdd(&gsum[b * H2 + o], s);
    if (o == 0) atomicAdd(&gcnt[b], 1.f);
}

// ---------------------------------------------------------------- head MLP
__global__ void k_head(const float* __restrict__ gsum, const float* __restrict__ gcnt,
                       const float* __restrict__ fc1w, const float* __restrict__ fc1b,
                       const float* __restrict__ fc2w, const float* __restrict__ fc2b,
                       float* __restrict__ out) {
    int g = threadIdx.x;
    if (g >= NGRAPH) return;
    float gm[64], t1[64];
    float c = fmaxf(gcnt[g], 1.f);
    for (int i = 0; i < 64; ++i) gm[i] = gsum[g * 64 + i] / c;
    for (int o = 0; o < 64; ++o) {
        float s = fc1b[o];
        for (int i = 0; i < 64; ++i) s += gm[i] * fc1w[i * 64 + o];
        t1[o] = (s > 0.f) ? s : (expf(s) - 1.f);
    }
    for (int o = 0; o < 64; ++o) {
        float s = fc2b[o];
        for (int i = 0; i < 64; ++i) s += t1[i] * fc2w[i * 64 + o];
        out[g * 64 + o] = s;
    }
}

// ---------------------------------------------------------------- launch
extern "C" void kernel_launch(void* const* d_in, const int* in_sizes, int n_in,
                              void* d_out, int out_size, void* d_ws, size_t ws_size,
                              hipStream_t stream) {
    const float* x      = (const float*)d_in[0];
    const int*   ei     = (const int*)d_in[1];
    const float* ea     = (const float*)d_in[2];
    const int*   batch  = (const int*)d_in[3];
    const float* nn1_w1 = (const float*)d_in[4],  *nn1_b1 = (const float*)d_in[5];
    const float* nn1_w2 = (const float*)d_in[6],  *nn1_b2 = (const float*)d_in[7];
    const float* root1  = (const float*)d_in[8],  *bias1  = (const float*)d_in[9];
    const float* nn2_w1 = (const float*)d_in[10], *nn2_b1 = (const float*)d_in[11];
    const float* nn2_w2 = (const float*)d_in[12], *nn2_b2 = (const float*)d_in[13];
    const float* root2  = (const float*)d_in[14], *bias2  = (const float*)d_in[15];
    const float* fc1w   = (const float*)d_in[16], *fc1b   = (const float*)d_in[17];
    const float* fc2w   = (const float*)d_in[18], *fc2b   = (const float*)d_in[19];

    const int N = in_sizes[0] / NFEAT;
    const int E = in_sizes[1] / 2;
    const int* src = ei;
    const int* dst = ei + E;

    // ---- workspace carve-up (zero-needed buffers first, contiguous) ----
    char* w = (char*)d_ws;
    auto alloc = [&](size_t bytes) {
        char* p = w;
        w += (bytes + 255) & ~(size_t)255;
        return p;
    };
    float* agg1 = (float*)alloc((size_t)N * H1 * 4);
    float* agg2 = (float*)alloc((size_t)N * H2 * 4);
    float* deg  = (float*)alloc((size_t)N * 4);
    float* gsum = (float*)alloc((size_t)NGRAPH * 64 * 4);
    float* gcnt = (float*)alloc((size_t)NGRAPH * 4);
    char*  zend = w;
    _Float16* he1 = (_Float16*)alloc((size_t)E * 16 * 2);
    _Float16* he2 = (_Float16*)alloc((size_t)E * 16 * 2);
    _Float16* xf  = (_Float16*)alloc((size_t)N * 32 * 2);
    _Float16* hf  = (_Float16*)alloc((size_t)N * 32 * 2);
    float*    hn  = (float*)alloc((size_t)N * H1 * 4);
    _Float16* wp1 = (_Float16*)alloc((size_t)(H1 / 16) * KSTEPS * 512 * 2);
    _Float16* wp2 = (_Float16*)alloc((size_t)(H2 / 16) * KSTEPS * 512 * 2);

    int zn = (int)((zend - (char*)agg1) / 4);
    k_zero<<<(zn + 255) / 256, 256, 0, stream>>>(agg1, zn);

    k_edge_mlp<<<(E + 255) / 256, 256, 0, stream>>>(ea, dst, nn1_w1, nn1_b1,
                                                    nn2_w1, nn2_b1, he1, he2, deg, E);
    k_pack_x<<<(N * 32 + 255) / 256, 256, 0, stream>>>(x, xf, N);

    const int wn = ((H1 / 16) + (H2 / 16)) * KSTEPS * 512;
    k_pack_w<<<(wn + 255) / 256, 256, 0, stream>>>(nn1_w2, nn1_b2, nn2_w2, nn2_b2, wp1, wp2);

    const int waves  = (E + 15) / 16;
    const int blocks = (waves + 7) / 8;          // 8 waves (256 thr) per block
    k_edge_gemm<H1 / 16><<<blocks, 256, 0, stream>>>(xf, he1, wp1, src, dst, agg1, E);
    k_node1<<<(N * H1 + 255) / 256, 256, 0, stream>>>(x, agg1, deg, root1, bias1, hn, hf, N);
    k_edge_gemm<H2 / 16><<<blocks, 256, 0, stream>>>(hf, he2, wp2, src, dst, agg2, E);

    float* out = (float*)d_out;
    k_node2<<<(N * H2 + 255) / 256, 256, 0, stream>>>(hn, agg2, deg, batch, root2, bias2,
                                                      out + NGRAPH * 64, gsum, gcnt, N);
    k_head<<<1, 64, 0, stream>>>(gsum, gcnt, fc1w, fc1b, fc2w, fc2b, out);
}
